// TtT5Attention_46540265620169
// MI455X (gfx1250) — compile-verified, matrix operand order checked by reading, not run on hardware
//
#include <hip/hip_runtime.h>
#include <hip/hip_bf16.h>

// ---------------------------------------------------------------------------
// T5 attention for MI455X (gfx1250, wave32, WMMA 16x16x32 bf16).
//   B=1, S=2048, D_MODEL=1024, H=16, Dk=64, NUM_BUCKETS=32, MAX_DIST=128
// bf16 matmuls, f32 accumulation (v_wmma_f32_16x16x32_bf16).
// Flash attention with kv-block 64; double-buffered LDS tiles staged with
// GLOBAL_LOAD_ASYNC_TO_LDS_B128 (ASYNCcnt) so the async engine fills the
// next tile while WMMAs run on the current one. One barrier per k-step.
// ---------------------------------------------------------------------------

typedef __bf16 bf16_t;
typedef __attribute__((ext_vector_type(16))) __bf16 v16bf;
typedef __attribute__((ext_vector_type(8)))  __bf16 v8bf;
typedef __attribute__((ext_vector_type(8)))  float  v8f;

union Frag {
    v16bf v;
    v8bf  h[2];
};

#define S_LEN   2048
#define DMODEL  1024
#define NHEADS  16
#define DKV     64

#if defined(__HIP_DEVICE_COMPILE__) && __has_builtin(__builtin_amdgcn_global_load_async_to_lds_b128)
#define HAS_ASYNC_LDS 1
#else
#define HAS_ASYNC_LDS 0
#endif

#if HAS_ASYNC_LDS
typedef int v4i_vs __attribute__((vector_size(16)));
typedef __attribute__((address_space(1))) v4i_vs as1_v4i;
typedef __attribute__((address_space(3))) v4i_vs as3_v4i;
#endif

// copy 8 bf16 (16 bytes) global -> LDS (async when available)
__device__ __forceinline__ void stage16B(const bf16_t* __restrict__ g, bf16_t* l) {
#if HAS_ASYNC_LDS
    __builtin_amdgcn_global_load_async_to_lds_b128((as1_v4i*)g, (as3_v4i*)l, 0, 0);
#else
    *(v8bf*)l = *(const v8bf*)g;
#endif
}

__device__ __forceinline__ void stage_wait() {
#if HAS_ASYNC_LDS
#if __has_builtin(__builtin_amdgcn_s_wait_asynccnt)
    __builtin_amdgcn_s_wait_asynccnt(0);
#else
    asm volatile("s_wait_asynccnt 0x0" ::: "memory");
#endif
#endif
}

// load two 16x32 B fragments (kc=0/1) from one 64-half LDS row (base incl. koff)
__device__ __forceinline__ void load_frag2(Frag& f0, Frag& f1, const bf16_t* base) {
    f0.h[0] = *(const v8bf*)(base + 0);
    f0.h[1] = *(const v8bf*)(base + 16);
    f1.h[0] = *(const v8bf*)(base + 32);
    f1.h[1] = *(const v8bf*)(base + 48);
}

#define WMMA_BF16(A_, B_, C_) \
    __builtin_amdgcn_wmma_f32_16x16x32_bf16(false, (A_).v, false, (B_).v, (short)0, (C_), false, false)

// ---------------------------------------------------------------------------
// f32 -> bf16 conversion (n multiple of 8)
// ---------------------------------------------------------------------------
__global__ __launch_bounds__(256)
void f32_to_bf16_kernel(const float* __restrict__ in, bf16_t* __restrict__ out, int n) {
    int i = (blockIdx.x * blockDim.x + threadIdx.x) * 8;
    if (i + 7 < n) {
        float4 f0 = *(const float4*)(in + i);
        float4 f1 = *(const float4*)(in + i + 4);
        v8bf o;
        o[0] = (bf16_t)f0.x; o[1] = (bf16_t)f0.y; o[2] = (bf16_t)f0.z; o[3] = (bf16_t)f0.w;
        o[4] = (bf16_t)f1.x; o[5] = (bf16_t)f1.y; o[6] = (bf16_t)f1.z; o[7] = (bf16_t)f1.w;
        *(v8bf*)(out + i) = o;
    }
}

// ---------------------------------------------------------------------------
// T5 relative-position bias table: biasT[h][d + 2047] for d = k - q
// ---------------------------------------------------------------------------
__global__ __launch_bounds__(256)
void bias_table_kernel(const float* __restrict__ rel_bias, float* __restrict__ biasT) {
    int t = blockIdx.x * blockDim.x + threadIdx.x;
    if (t >= NHEADS * 4095) return;
    int h = t / 4095;
    int d = t % 4095 - 2047;           // memory_pos - query_pos
    int base = (d > 0) ? 16 : 0;
    int ad = d < 0 ? -d : d;
    int b;
    if (ad < 8) {
        b = ad;
    } else {
        float lf = logf((float)ad * (1.0f / 8.0f)) * (8.0f / logf(16.0f));
        int large = 8 + (int)lf;
        b = large < 15 ? large : 15;
    }
    biasT[t] = rel_bias[(base + b) * NHEADS + h];
}

// ---------------------------------------------------------------------------
// GEMM:  out[M][N] = A[M][K] @ W[N][K]^T   (bf16 in, f32 accumulate)
// 256 threads = 8 waves; each wave: 16x128 C strip. k-step 64, double-
// buffered async W tile (128x64 = 16KB each). One barrier per k-step.
// ---------------------------------------------------------------------------
template <bool SPLIT_HEADS>
__global__ __launch_bounds__(256)
void gemm_bf16_kernel(const bf16_t* __restrict__ A, const bf16_t* __restrict__ W,
                      void* __restrict__ outp, int M, int N, int K) {
    __shared__ bf16_t Wt[2][128 * 64];

    const int tid    = threadIdx.x;
    const int wave   = tid >> 5;
    const int lane   = tid & 31;
    const int lo     = lane & 15;
    const int hiHalf = lane >> 4;
    const int koff   = hiHalf * 8;
    const int rb = blockIdx.x * 128;
    const int cb = blockIdx.y * 128;
    const int r0 = rb + wave * 16;

    v8f c[8];
#pragma unroll
    for (int nt = 0; nt < 8; ++nt) c[nt] = (v8f){};

    const int e  = tid * 32;               // staging: 32 halves per thread
    const int wr = e >> 6;                 // 0..127
    const int wc = e & 63;                 // 0 or 32
    const bf16_t* arow = A + (size_t)(r0 + lo) * K;

    // prologue: stage k-block 0, load A fragments for k-block 0
    {
        const bf16_t* wsrc = W + (size_t)(cb + wr) * K + wc;
        bf16_t* dst = &Wt[0][e];
        stage16B(wsrc,      dst);      stage16B(wsrc + 8,  dst + 8);
        stage16B(wsrc + 16, dst + 16); stage16B(wsrc + 24, dst + 24);
    }
    Frag a0, a1, na0, na1;
    load_frag2(a0, a1, arow + koff);

    for (int kb = 0; kb < K; kb += 64) {
        const int cur = (kb >> 6) & 1;
        stage_wait();
        __syncthreads();

        if (kb + 64 < K) {                 // stage next tile while computing
            const bf16_t* wsrc = W + (size_t)(cb + wr) * K + kb + 64 + wc;
            bf16_t* dst = &Wt[1 - cur][e];
            stage16B(wsrc,      dst);      stage16B(wsrc + 8,  dst + 8);
            stage16B(wsrc + 16, dst + 16); stage16B(wsrc + 24, dst + 24);
            load_frag2(na0, na1, arow + kb + 64 + koff);
        }

#pragma unroll
        for (int g = 0; g < 4; ++g) {      // two N-tiles per burst
            Frag b00, b01, b10, b11;
            load_frag2(b00, b01, &Wt[cur][((2 * g)     * 16 + lo) * 64 + koff]);
            load_frag2(b10, b11, &Wt[cur][((2 * g + 1) * 16 + lo) * 64 + koff]);
            c[2 * g]     = WMMA_BF16(a0, b00, c[2 * g]);
            c[2 * g]     = WMMA_BF16(a1, b01, c[2 * g]);
            c[2 * g + 1] = WMMA_BF16(a0, b10, c[2 * g + 1]);
            c[2 * g + 1] = WMMA_BF16(a1, b11, c[2 * g + 1]);
        }
        a0 = na0;
        a1 = na1;
    }

#pragma unroll
    for (int nt = 0; nt < 8; ++nt) {
#pragma unroll
        for (int r = 0; r < 8; ++r) {
            const int row = r0 + r + 8 * hiHalf;
            const int col = cb + nt * 16 + lo;
            if (SPLIT_HEADS) {
                const int hh = col >> 6, d = col & 63;
                ((bf16_t*)outp)[((size_t)hh * M + row) * DKV + d] = (bf16_t)c[nt][r];
            } else {
                ((float*)outp)[(size_t)row * N + col] = c[nt][r];
            }
        }
    }
}

// ---------------------------------------------------------------------------
// Flash attention: one head per blockIdx.y, 128 query rows per block,
// 8 waves x 16 query rows, kv blocks of 64. Double-buffered K (async) and
// V-transposed LDS tiles; one barrier per kv-step.
// ---------------------------------------------------------------------------
__global__ __launch_bounds__(256)
void attn_kernel(const bf16_t* __restrict__ Qh, const bf16_t* __restrict__ Kh,
                 const bf16_t* __restrict__ Vh, const float* __restrict__ biasT,
                 bf16_t* __restrict__ ctx) {
    __shared__ bf16_t Kt[2][64 * 64];         // K blocks, row-major [kv][d]  (2x8KB)
    __shared__ bf16_t VT[2][64 * 64];         // V blocks transposed [d][kv]  (2x8KB)
    __shared__ bf16_t Pst[8 * 16 * 64];       // per-wave P staging           (16KB)

    const int tid    = threadIdx.x;
    const int wave   = tid >> 5;
    const int lane   = tid & 31;
    const int lo     = lane & 15;
    const int hiHalf = lane >> 4;
    const int koff   = hiHalf * 8;
    const int h  = blockIdx.y;
    const int q0 = blockIdx.x * 128 + wave * 16;

    const bf16_t* Qb = Qh + (size_t)h * S_LEN * DKV;
    const bf16_t* Kb = Kh + (size_t)h * S_LEN * DKV;
    const bf16_t* Vb = Vh + (size_t)h * S_LEN * DKV;
    const float*  bT = biasT + h * 4095 + 2047;

    // Q fragments (K-dim 64 -> two 16x32 chunks)
    Frag aq0, aq1;
    load_frag2(aq0, aq1, Qb + (size_t)(q0 + lo) * DKV + koff);

    v8f o[4];
#pragma unroll
    for (int nt = 0; nt < 4; ++nt) o[nt] = (v8f){};
    float m[8], l[8];
#pragma unroll
    for (int r = 0; r < 8; ++r) { m[r] = -3.0e38f; l[r] = 0.0f; }

    bf16_t* Pw = Pst + wave * 16 * 64;

    const int e   = tid * 16;                 // staging: 16 halves per thread
    const int vr  = e >> 6;                   // 0..63
    const int vc0 = e & 63;                   // {0,16,32,48}

    // prologue: async K block 0, V block 0 into registers
    {
        const bf16_t* ksrc = Kb + e;
        stage16B(ksrc,     &Kt[0][e]);
        stage16B(ksrc + 8, &Kt[0][e + 8]);
    }
    v8bf v0 = *(const v8bf*)(Vb + (size_t)vr * DKV + vc0);
    v8bf v1 = *(const v8bf*)(Vb + (size_t)vr * DKV + vc0 + 8);

    for (int kb = 0; kb < S_LEN; kb += 64) {
        const int cur = (kb >> 6) & 1;

        // commit V regs (block kb) into VT[cur]
#pragma unroll
        for (int i = 0; i < 8; ++i) {
            VT[cur][(vc0 + i) * 64 + vr]     = v0[i];
            VT[cur][(vc0 + 8 + i) * 64 + vr] = v1[i];
        }
        stage_wait();                         // K block kb landed in Kt[cur]
        __syncthreads();

        if (kb + 64 < S_LEN) {                // stage next kv block while computing
            const bf16_t* ksrc = Kb + (size_t)(kb + 64) * DKV + e;
            stage16B(ksrc,     &Kt[1 - cur][e]);
            stage16B(ksrc + 8, &Kt[1 - cur][e + 8]);
            const bf16_t* vsrc = Vb + (size_t)(kb + 64 + vr) * DKV + vc0;
            v0 = *(const v8bf*)(vsrc);
            v1 = *(const v8bf*)(vsrc + 8);
        }

        // ---- S = Q @ K^T : four 16x16 tiles ----
        v8f s[4];
#pragma unroll
        for (int g = 0; g < 2; ++g) {
            Frag b00, b01, b10, b11;
            load_frag2(b00, b01, &Kt[cur][((2 * g)     * 16 + lo) * 64 + koff]);
            load_frag2(b10, b11, &Kt[cur][((2 * g + 1) * 16 + lo) * 64 + koff]);
            s[2 * g]     = WMMA_BF16(aq0, b00, (v8f){});
            s[2 * g]     = WMMA_BF16(aq1, b01, s[2 * g]);
            s[2 * g + 1] = WMMA_BF16(aq0, b10, (v8f){});
            s[2 * g + 1] = WMMA_BF16(aq1, b11, s[2 * g + 1]);
        }

        // ---- bias + online softmax (fold 4 tiles, butterfly once) ----
        float t[8];
#pragma unroll
        for (int r = 0; r < 8; ++r) {
            const int qrow = q0 + r + 8 * hiHalf;
            const int c0i  = kb + lo - qrow;
            s[0][r] += bT[c0i];
            s[1][r] += bT[c0i + 16];
            s[2][r] += bT[c0i + 32];
            s[3][r] += bT[c0i + 48];
            t[r] = fmaxf(fmaxf(s[0][r], s[1][r]), fmaxf(s[2][r], s[3][r]));
        }
#pragma unroll
        for (int mask = 1; mask < 16; mask <<= 1)
#pragma unroll
            for (int r = 0; r < 8; ++r)
                t[r] = fmaxf(t[r], __shfl_xor(t[r], mask, 32));

        float rs[8];
#pragma unroll
        for (int r = 0; r < 8; ++r) {
            const float mn = fmaxf(m[r], t[r]);
            const float sc = __expf(m[r] - mn);
            m[r] = mn;
            float p0 = __expf(s[0][r] - mn);
            float p1 = __expf(s[1][r] - mn);
            float p2 = __expf(s[2][r] - mn);
            float p3 = __expf(s[3][r] - mn);
            s[0][r] = p0; s[1][r] = p1; s[2][r] = p2; s[3][r] = p3;
            rs[r] = (p0 + p1) + (p2 + p3);
            l[r] *= sc;
            o[0][r] *= sc; o[1][r] *= sc; o[2][r] *= sc; o[3][r] *= sc;
        }
#pragma unroll
        for (int mask = 1; mask < 16; mask <<= 1)
#pragma unroll
            for (int r = 0; r < 8; ++r)
                rs[r] += __shfl_xor(rs[r], mask, 32);
#pragma unroll
        for (int r = 0; r < 8; ++r) l[r] += rs[r];

        // ---- D-layout -> A-layout transpose of P through per-wave LDS ----
#pragma unroll
        for (int r = 0; r < 8; ++r) {
            const int rloc = r + 8 * hiHalf;
            Pw[rloc * 64 + lo]      = (bf16_t)s[0][r];
            Pw[rloc * 64 + 16 + lo] = (bf16_t)s[1][r];
            Pw[rloc * 64 + 32 + lo] = (bf16_t)s[2][r];
            Pw[rloc * 64 + 48 + lo] = (bf16_t)s[3][r];
        }
        Frag ap0, ap1;
        load_frag2(ap0, ap1, &Pw[lo * 64 + koff]);

        // ---- O += P @ V ----
#pragma unroll
        for (int g = 0; g < 2; ++g) {
            Frag b00, b01, b10, b11;
            load_frag2(b00, b01, &VT[cur][((2 * g)     * 16 + lo) * 64 + koff]);
            load_frag2(b10, b11, &VT[cur][((2 * g + 1) * 16 + lo) * 64 + koff]);
            o[2 * g]     = WMMA_BF16(ap0, b00, o[2 * g]);
            o[2 * g]     = WMMA_BF16(ap1, b01, o[2 * g]);
            o[2 * g + 1] = WMMA_BF16(ap0, b10, o[2 * g + 1]);
            o[2 * g + 1] = WMMA_BF16(ap1, b11, o[2 * g + 1]);
        }
    }

    // ---- epilogue: ctx[row][h*64 + col] = O / l ----
#pragma unroll
    for (int nt = 0; nt < 4; ++nt) {
#pragma unroll
        for (int r = 0; r < 8; ++r) {
            const int row = q0 + r + 8 * hiHalf;
            const int col = h * DKV + nt * 16 + lo;
            ctx[(size_t)row * DMODEL + col] = (bf16_t)(o[nt][r] / l[r]);
        }
    }
}

// ---------------------------------------------------------------------------
// Host-side launch
// ---------------------------------------------------------------------------
extern "C" void kernel_launch(void* const* d_in, const int* in_sizes, int n_in,
                              void* d_out, int out_size, void* d_ws, size_t ws_size,
                              hipStream_t stream) {
    (void)in_sizes; (void)n_in; (void)out_size; (void)ws_size;

    const float* X    = (const float*)d_in[0];
    const float* wq   = (const float*)d_in[1];
    const float* wk   = (const float*)d_in[2];
    const float* wv   = (const float*)d_in[3];
    const float* wo   = (const float*)d_in[4];
    const float* relb = (const float*)d_in[5];
    float* out = (float*)d_out;

    char*  ws  = (char*)d_ws;
    size_t off = 0;
    auto carve = [&](size_t bytes) -> char* {
        char* p = ws + off;
        off = (off + bytes + 255) & ~(size_t)255;
        return p;
    };

    bf16_t* Xbf  = (bf16_t*)carve((size_t)S_LEN * DMODEL * 2);
    bf16_t* Wqb  = (bf16_t*)carve((size_t)DMODEL * DMODEL * 2);
    bf16_t* Wkb  = (bf16_t*)carve((size_t)DMODEL * DMODEL * 2);
    bf16_t* Wvb  = (bf16_t*)carve((size_t)DMODEL * DMODEL * 2);
    bf16_t* Wob  = (bf16_t*)carve((size_t)DMODEL * DMODEL * 2);
    bf16_t* Qh   = (bf16_t*)carve((size_t)S_LEN * DMODEL * 2);   // [16][2048][64]
    bf16_t* Kh   = (bf16_t*)carve((size_t)S_LEN * DMODEL * 2);
    bf16_t* Vh   = (bf16_t*)carve((size_t)S_LEN * DMODEL * 2);
    bf16_t* ctx  = (bf16_t*)carve((size_t)S_LEN * DMODEL * 2);
    float*  bias = (float*)carve((size_t)NHEADS * 4095 * 4);

    // 1) convert inputs to bf16
    f32_to_bf16_kernel<<<(S_LEN * DMODEL) / 2048, 256, 0, stream>>>(X, Xbf, S_LEN * DMODEL);
    f32_to_bf16_kernel<<<(DMODEL * DMODEL) / 2048, 256, 0, stream>>>(wq, Wqb, DMODEL * DMODEL);
    f32_to_bf16_kernel<<<(DMODEL * DMODEL) / 2048, 256, 0, stream>>>(wk, Wkb, DMODEL * DMODEL);
    f32_to_bf16_kernel<<<(DMODEL * DMODEL) / 2048, 256, 0, stream>>>(wv, Wvb, DMODEL * DMODEL);
    f32_to_bf16_kernel<<<(DMODEL * DMODEL) / 2048, 256, 0, stream>>>(wo, Wob, DMODEL * DMODEL);

    // 2) bias table
    bias_table_kernel<<<(NHEADS * 4095 + 255) / 256, 256, 0, stream>>>(relb, bias);

    // 3) Q/K/V projections, written per-head [h][s][64]
    dim3 ggrid(S_LEN / 128, DMODEL / 128);
    gemm_bf16_kernel<true><<<ggrid, 256, 0, stream>>>(Xbf, Wqb, Qh, S_LEN, DMODEL, DMODEL);
    gemm_bf16_kernel<true><<<ggrid, 256, 0, stream>>>(Xbf, Wkb, Kh, S_LEN, DMODEL, DMODEL);
    gemm_bf16_kernel<true><<<ggrid, 256, 0, stream>>>(Xbf, Wvb, Vh, S_LEN, DMODEL, DMODEL);

    // 4) flash attention with relative bias
    dim3 agrid(S_LEN / 128, NHEADS);
    attn_kernel<<<agrid, 256, 0, stream>>>(Qh, Kh, Vh, bias, ctx);

    // 5) output projection -> f32 d_out
    gemm_bf16_kernel<false><<<ggrid, 256, 0, stream>>>(ctx, Wob, out, S_LEN, DMODEL, DMODEL);
}
